// ResidualVectorQuantizer_86741159510609
// MI455X (gfx1250) — matrix operand order repaired and checked by reading, not compiled, not run on hardware
//
#include <hip/hip_runtime.h>

typedef __attribute__((ext_vector_type(16))) __bf16          v16bf;
typedef __attribute__((ext_vector_type(16))) unsigned short  v16u16;
typedef __attribute__((ext_vector_type(8)))  float           v8f;

#define QSTAGES 4
#define KCODES  256
#define DDIM    32
#define WAVES   8
#define TILES_PER_WAVE 2
#define ROWS_PER_BLOCK 256   // 8 waves x 2 tiles x 16 rows

__device__ __forceinline__ unsigned bf16rne(float f) {
  unsigned u = __float_as_uint(f);
  return (u + 0x7FFFu + ((u >> 16) & 1u)) >> 16;   // round-to-nearest-even to bf16
}
__device__ __forceinline__ float bf16tof(unsigned h) { return __uint_as_float(h << 16); }

// ---------------------------------------------------------------------------
// Prep: split codebooks into bf16 hi/lo and compute per-code squared norms.
// One thread per code (Q*K = 1024 threads).
// ---------------------------------------------------------------------------
__global__ void rvq_prep(const float* __restrict__ cb,
                         unsigned short* __restrict__ wsHi,
                         unsigned short* __restrict__ wsLo,
                         float* __restrict__ wsNorm) {
  int code = blockIdx.x * blockDim.x + threadIdx.x;
  if (code >= QSTAGES * KCODES) return;
  const float* src = cb + (size_t)code * DDIM;
  float nrm = 0.f;
#pragma unroll
  for (int j = 0; j < DDIM; ++j) {
    float f = src[j];
    nrm += f * f;
    unsigned h = bf16rne(f);
    wsHi[code * DDIM + j] = (unsigned short)h;
    wsLo[code * DDIM + j] = (unsigned short)bf16rne(f - bf16tof(h));
  }
  wsNorm[code] = nrm;
}

// ---------------------------------------------------------------------------
// Main RVQ kernel. 256 threads = 8 wave32; each wave owns two 16-row tiles
// (dual independent WMMA accumulation chains sharing the B fragments).
// ---------------------------------------------------------------------------
__global__ __launch_bounds__(256) void rvq_main(
    const float* __restrict__ x, const float* __restrict__ cb,
    float* __restrict__ out,
    const unsigned short* __restrict__ wsHi,
    const unsigned short* __restrict__ wsLo,
    const float* __restrict__ wsNorm,
    float* __restrict__ partials, int N) {
  __shared__ float s_res[WAVES][TILES_PER_WAVE][16][DDIM];    // 32 KB residuals
  __shared__ alignas(32) unsigned short s_hi[KCODES * DDIM];  // 16 KB
  __shared__ alignas(32) unsigned short s_lo[KCODES * DDIM];  // 16 KB
  __shared__ float s_norm[KCODES];                            // 1 KB
  __shared__ int   s_idx[WAVES][TILES_PER_WAVE][16];
  __shared__ float s_lsum[WAVES];

  const int t    = threadIdx.x;
  const int lane = t & 31;
  const int w    = t >> 5;
  const int hsel = lane >> 4;    // which half of the wave (lane group)
  const int l16  = lane & 15;
  const int rowBase  = blockIdx.x * ROWS_PER_BLOCK;
  const int waveRow0 = rowBase + w * (TILES_PER_WAVE * 16);

  // load initial residual (lane -> row l16 of each tile, dims hsel*16..+15)
#pragma unroll
  for (int p = 0; p < TILES_PER_WAVE; ++p) {
    int rowG = waveRow0 + p * 16 + l16;
    const float4* xp = (const float4*)(x + (size_t)rowG * DDIM + hsel * 16);
    float4* rp = (float4*)&s_res[w][p][l16][hsel * 16];
#pragma unroll
    for (int i = 0; i < 4; ++i) rp[i] = xp[i];
  }

  float loss_acc = 0.f;
  float* outIdx = out + (size_t)N * DDIM + 1;   // indices region (after x_q, loss)

  for (int q = 0; q < QSTAGES; ++q) {
    __syncthreads();
    // stage codebook -> LDS (16KB hi + 16KB lo + 1KB norms), cooperative
    {
      const uint4* srcH = (const uint4*)(wsHi + (q << 13));
      const uint4* srcL = (const uint4*)(wsLo + (q << 13));
      uint4* dH = (uint4*)s_hi;
      uint4* dL = (uint4*)s_lo;
#pragma unroll
      for (int i = 0; i < 4; ++i) {
        dH[t + 256 * i] = srcH[t + 256 * i];
        dL[t + 256 * i] = srcL[t + 256 * i];
      }
      s_norm[t] = wsNorm[(q << 8) + t];
    }
    __syncthreads();

    // prefetch next stage's codebook lines into cache while we compute
    if (q + 1 < QSTAGES) {
      const char* ph = (const char*)(wsHi + ((q + 1) << 13));
      const char* pl = (const char*)(wsLo + ((q + 1) << 13));
      const char* p = (t < 128) ? ph + (size_t)t * 128 : pl + (size_t)(t - 128) * 128;
      __builtin_prefetch(p, 0, 0);
    }

    // A fragments per tile: 16x32 bf16 hi/lo split of residual rows.
    // Layout: lane row = l16; VGPR v holds K = hsel*16 + 2*v .. +1
    v16bf ah[TILES_PER_WAVE], al[TILES_PER_WAVE];
#pragma unroll
    for (int p = 0; p < TILES_PER_WAVE; ++p) {
      v16u16 ahU, alU;
      const float* rrow = &s_res[w][p][l16][0];
#pragma unroll
      for (int v = 0; v < 8; ++v) {
        int kb = (hsel << 4) + (v << 1);
        float f0 = rrow[kb], f1 = rrow[kb + 1];
        unsigned h0 = bf16rne(f0), h1 = bf16rne(f1);
        ahU[2 * v]     = (unsigned short)h0;
        ahU[2 * v + 1] = (unsigned short)h1;
        alU[2 * v]     = (unsigned short)bf16rne(f0 - bf16tof(h0));
        alU[2 * v + 1] = (unsigned short)bf16rne(f1 - bf16tof(h1));
      }
      ah[p] = __builtin_bit_cast(v16bf, ahU);
      al[p] = __builtin_bit_cast(v16bf, alU);
    }

    float best[TILES_PER_WAVE][8];
    int   bidx[TILES_PER_WAVE][8];
#pragma unroll
    for (int p = 0; p < TILES_PER_WAVE; ++p)
#pragma unroll
      for (int v = 0; v < 8; ++v) { best[p][v] = 3.4e38f; bidx[p][v] = 0; }

    // 16 column tiles; each B hi/lo pair feeds 6 WMMAs (2 independent chains)
#pragma unroll 2
    for (int n0 = 0; n0 < KCODES; n0 += 16) {
      // B layout: column = l16, lanes 0-15 hold K 0-15, lanes 16-31 hold K 16-31
      v16u16 bhU = *(const v16u16*)&s_hi[(n0 + l16) * DDIM + hsel * 16];
      v16u16 blU = *(const v16u16*)&s_lo[(n0 + l16) * DDIM + hsel * 16];
      v16bf bh = __builtin_bit_cast(v16bf, bhU);
      v16bf bl = __builtin_bit_cast(v16bf, blU);
      v8f a0 = {}, a1 = {};
      // r.e ~= rh.eh + rl.eh + rh.el (fp32-grade), interleaved chains
      a0 = __builtin_amdgcn_wmma_f32_16x16x32_bf16(false, ah[0], false, bh, (short)0, a0, false, false);
      a1 = __builtin_amdgcn_wmma_f32_16x16x32_bf16(false, ah[1], false, bh, (short)0, a1, false, false);
      a0 = __builtin_amdgcn_wmma_f32_16x16x32_bf16(false, al[0], false, bh, (short)0, a0, false, false);
      a1 = __builtin_amdgcn_wmma_f32_16x16x32_bf16(false, al[1], false, bh, (short)0, a1, false, false);
      a0 = __builtin_amdgcn_wmma_f32_16x16x32_bf16(false, ah[0], false, bl, (short)0, a0, false, false);
      a1 = __builtin_amdgcn_wmma_f32_16x16x32_bf16(false, ah[1], false, bl, (short)0, a1, false, false);
      float nrm = s_norm[n0 + l16];
      int cand = n0 + l16;
#pragma unroll
      for (int v = 0; v < 8; ++v) {
        float s0 = fmaf(-2.f, a0[v], nrm);   // ||e||^2 - 2 r.e
        if (s0 < best[0][v]) { best[0][v] = s0; bidx[0][v] = cand; }
        float s1 = fmaf(-2.f, a1[v], nrm);
        if (s1 < best[1][v]) { best[1][v] = s1; bidx[1][v] = cand; }
      }
    }

    // argmin across the 16-lane half group (first-index tie-break, like argmin)
#pragma unroll
    for (int m = 1; m < 16; m <<= 1) {
#pragma unroll
      for (int p = 0; p < TILES_PER_WAVE; ++p)
#pragma unroll
        for (int v = 0; v < 8; ++v) {
          float ov = __shfl_xor(best[p][v], m, 32);
          int   oi = __shfl_xor(bidx[p][v], m, 32);
          if (ov < best[p][v] || (ov == best[p][v] && oi < bidx[p][v])) {
            best[p][v] = ov; bidx[p][v] = oi;
          }
        }
    }

    if (l16 == 0) {
#pragma unroll
      for (int v = 0; v < 8; ++v) {
        int r = hsel * 8 + v;   // D row M = v + 8*hsel
        s_idx[w][0][r] = bidx[0][v];
        s_idx[w][1][r] = bidx[1][v];
        outIdx[(size_t)(waveRow0 + r) * QSTAGES + q]      = (float)bidx[0][v];
        outIdx[(size_t)(waveRow0 + 16 + r) * QSTAGES + q] = (float)bidx[1][v];
      }
    }
    __syncthreads();

    // exact fp32 update: gather e, loss += (e-r)^2, residual -= e
#pragma unroll
    for (int p = 0; p < TILES_PER_WAVE; ++p) {
      int idxr = s_idx[w][p][l16];
      const float4* ep = (const float4*)(cb + ((q << 13) + (idxr << 5) + (hsel << 4)));
      float4* rp = (float4*)&s_res[w][p][l16][hsel * 16];
#pragma unroll
      for (int i = 0; i < 4; ++i) {
        float4 e = ep[i];
        float4 r = rp[i];
        float d0 = e.x - r.x, d1 = e.y - r.y, d2 = e.z - r.z, d3 = e.w - r.w;
        loss_acc += d0 * d0 + d1 * d1 + d2 * d2 + d3 * d3;
        float4 nr; nr.x = -d0; nr.y = -d1; nr.z = -d2; nr.w = -d3;  // r - e
        rp[i] = nr;
      }
    }
  }

  // x_q = x - residual_final (exact: residual_final = x - sum(e) elementwise)
#pragma unroll
  for (int p = 0; p < TILES_PER_WAVE; ++p) {
    int rowG = waveRow0 + p * 16 + l16;
    const float4* xp = (const float4*)(x + (size_t)rowG * DDIM + hsel * 16);
    const float4* rp = (const float4*)&s_res[w][p][l16][hsel * 16];
    float4* op = (float4*)(out + (size_t)rowG * DDIM + hsel * 16);
#pragma unroll
    for (int i = 0; i < 4; ++i) {
      float4 xv = xp[i];
      float4 rv = rp[i];
      float4 o;
      o.x = xv.x - rv.x; o.y = xv.y - rv.y; o.z = xv.z - rv.z; o.w = xv.w - rv.w;
      op[i] = o;
    }
  }

  // deterministic loss partial per block
#pragma unroll
  for (int m = 16; m >= 1; m >>= 1) loss_acc += __shfl_xor(loss_acc, m, 32);
  if (lane == 0) s_lsum[w] = loss_acc;
  __syncthreads();
  if (t == 0) {
    float s = 0.f;
#pragma unroll
    for (int i = 0; i < WAVES; ++i) s += s_lsum[i];
    partials[blockIdx.x] = s;
  }
}

// ---------------------------------------------------------------------------
// Deterministic final loss reduction (single block, fixed order).
// ---------------------------------------------------------------------------
__global__ void rvq_loss_reduce(const float* __restrict__ partials,
                                float* __restrict__ out_loss, int n, float scale) {
  __shared__ float sm[256];
  float t = 0.f;
  for (int i = threadIdx.x; i < n; i += 256) t += partials[i];
  sm[threadIdx.x] = t;
  __syncthreads();
  for (int s = 128; s > 0; s >>= 1) {
    if ((int)threadIdx.x < s) sm[threadIdx.x] += sm[threadIdx.x + s];
    __syncthreads();
  }
  if (threadIdx.x == 0) out_loss[0] = sm[0] * scale;
}

extern "C" void kernel_launch(void* const* d_in, const int* in_sizes, int n_in,
                              void* d_out, int out_size, void* d_ws, size_t ws_size,
                              hipStream_t stream) {
  const float* x  = (const float*)d_in[0];        // [N, 32]
  const float* cb = (const float*)d_in[1];        // [4, 256, 32]
  float* out = (float*)d_out;                     // x_q (N*32) | loss (1) | indices (N*4)
  const int N = in_sizes[0] / DDIM;
  const int nBlocks = N / ROWS_PER_BLOCK;

  unsigned short* wsHi = (unsigned short*)d_ws;                 // 64 KB
  unsigned short* wsLo = wsHi + QSTAGES * KCODES * DDIM;        // 64 KB
  float* wsNorm   = (float*)(wsLo + QSTAGES * KCODES * DDIM);   // 4 KB
  float* partials = wsNorm + QSTAGES * KCODES;                  // nBlocks floats

  rvq_prep<<<QSTAGES, 256, 0, stream>>>(cb, wsHi, wsLo, wsNorm);
  rvq_main<<<nBlocks, 256, 0, stream>>>(x, cb, out, wsHi, wsLo, wsNorm, partials, N);
  const float scale = 1.25f / ((float)N * (float)DDIM * (float)QSTAGES);
  rvq_loss_reduce<<<1, 256, 0, stream>>>(partials, out + (size_t)N * DDIM, nBlocks, scale);
}
